// TtFalconAttentionPrefill_7559142441199
// MI455X (gfx1250) — matrix operand order repaired and checked by reading, not compiled, not run on hardware
//
#include <hip/hip_runtime.h>
#include <hip/hip_bf16.h>

// ---------------------------------------------------------------------------
// Falcon MQA prefill for MI455X (gfx1250, wave32, WMMA, TDM).
// Pipeline: f32->bf16 converts, QKV GEMM (bf16 WMMA + tensor_load_to_lds),
//           RoPE/split, flash-attention (bf16 WMMA, online softmax),
//           dense GEMM.
// Dims: SEQ=2048, HIDDEN=4544 (=71*64), QKV_N=4672 (=73*64), K=4544=71*64.
// ---------------------------------------------------------------------------

#define SEQ     2048
#define HIDDEN  4544
#define NHEADS  71
#define HDIM    64
#define QKV_N   ((NHEADS + 2) * HDIM)   // 4672

typedef __attribute__((ext_vector_type(16))) __bf16 v16bf;
typedef __attribute__((ext_vector_type(8)))  float  v8f;
typedef __attribute__((ext_vector_type(4)))  unsigned int v4u;
typedef __attribute__((ext_vector_type(8)))  int v8i;
typedef __attribute__((ext_vector_type(4)))  int v4i;

union BF16x16 { v16bf v; uint4 q[2]; };
union BF16x8  { uint4 q; __bf16 h[8]; };

#if defined(__has_builtin)
#if __has_builtin(__builtin_amdgcn_tensor_load_to_lds) && \
    __has_builtin(__builtin_amdgcn_s_wait_tensorcnt)
#define HAVE_TDM 1
#endif
#endif
#if __has_include(<hip/amd_detail/amd_gfx1250_TDM.h>)
#define TDM_6ARG 1
#endif

static __device__ __forceinline__ v8f wmma_bf16(v16bf a, v16bf b, v8f c) {
  return __builtin_amdgcn_wmma_f32_16x16x32_bf16(false, a, false, b,
                                                 (short)0, c, false, false);
}

static __device__ __forceinline__ v16bf load_frag(const __bf16* p0,
                                                  const __bf16* p1) {
  BF16x16 u;
  u.q[0] = *reinterpret_cast<const uint4*>(p0);
  u.q[1] = *reinterpret_cast<const uint4*>(p1);
  return u.v;
}

#if defined(HAVE_TDM)
// LDS byte offset of a __shared__ object (ptrtoint of addrspace(3) pointer).
static __device__ __forceinline__ unsigned lds_offset(const void* p) {
  return (unsigned)(unsigned long long)
      ((__attribute__((address_space(3))) const void*)p);
}

// 2D bf16 tile DMA: global (row-major, stride_d0 elems) -> LDS, with LDS
// row padding of 4 DWORDs after every 32 DWORDs (i.e. 64+8 bf16 rows).
// D# fields per CDNA5 ISA §8.3/8.4.
static __device__ __forceinline__ void tdm_load_tile_2d(
    unsigned lds_addr, const __bf16* gptr,
    unsigned tensor_d0, unsigned tensor_d1,
    unsigned tile_d0, unsigned tile_d1, unsigned stride_d0) {
  unsigned long long ga = (unsigned long long)(unsigned long long)(size_t)gptr;
  v4u g0;
  g0[0] = 1u;                                   // count=1, user descriptor
  g0[1] = lds_addr;                             // lds_addr (bytes)
  g0[2] = (unsigned)(ga & 0xFFFFFFFFu);         // global_addr[31:0]
  g0[3] = (unsigned)((ga >> 32) & 0x1FFFFFFu)   // global_addr[56:32]
          | (2u << 30);                         // type=2 ("image")
  const unsigned pad_en = 1u, pad_interval = 4u /*32 DW*/, pad_amt = 3u /*4 DW*/;
  v8i g1;
  g1[0] = (int)((1u << 16)                      // data_size=1 -> 2 bytes
                | (pad_en << 20) | (pad_interval << 22) | (pad_amt << 25));
  g1[1] = (int)((tensor_d0 & 0xFFFFu) << 16);   // tensor_dim0[15:0]
  g1[2] = (int)(((tensor_d0 >> 16) & 0xFFFFu) | ((tensor_d1 & 0xFFFFu) << 16));
  g1[3] = (int)(((tensor_d1 >> 16) & 0xFFFFu) | ((tile_d0 & 0xFFFFu) << 16));
  g1[4] = (int)(tile_d1 & 0xFFFFu);             // tile_dim1 (tile_dim2=0)
  g1[5] = (int)stride_d0;                       // tensor_dim0_stride[31:0]
  g1[6] = 0;                                    // stride hi / dim1 stride
  g1[7] = 0;
  v4i z4 = {0, 0, 0, 0};
#if defined(TDM_6ARG)
  v8i z8 = {0, 0, 0, 0, 0, 0, 0, 0};
  __builtin_amdgcn_tensor_load_to_lds(g0, g1, z4, z4, z8, 0);
#else
  __builtin_amdgcn_tensor_load_to_lds(g0, g1, z4, z4, 0);
#endif
}
#endif  // HAVE_TDM

// ---------------------------------------------------------------------------
// fp32 -> bf16 bulk convert (4 elems/thread)
// ---------------------------------------------------------------------------
__global__ void __launch_bounds__(256)
cvt_f32_bf16_v4(const float* __restrict__ in, __bf16* __restrict__ out,
                size_t n4) {
  size_t i = (size_t)blockIdx.x * blockDim.x + threadIdx.x;
  if (i >= n4) return;
  float4 f = *reinterpret_cast<const float4*>(in + i * 4);
  union { uint2 u; __bf16 h[4]; } o;
  o.h[0] = (__bf16)f.x; o.h[1] = (__bf16)f.y;
  o.h[2] = (__bf16)f.z; o.h[3] = (__bf16)f.w;
  *reinterpret_cast<uint2*>(out + i * 4) = o.u;
}

// ---------------------------------------------------------------------------
// C[M,N] (f32) = A[M,K] @ B[K,N], both bf16 row-major.
// Block: 256 thr = 8 waves; block tile 256x64; wave tile 64x32 (4x2 accums).
// K staged in 64-wide slabs: A via one TDM 2D tile-DMA per wave (32x64 rows
// into padded LDS rows), B transposed into LDS through VGPRs.
// Requires: M%256==0, N%64==0, K%64==0 (true for all uses here).
// ---------------------------------------------------------------------------
__global__ void __launch_bounds__(256)
gemm_bf16_f32(const __bf16* __restrict__ A, const __bf16* __restrict__ B,
              float* __restrict__ C, int M, int N, int K) {
  __shared__ __align__(16) __bf16 As[256][72];   // 256x64 (+8 pad) = 36 KB
  __shared__ __align__(16) __bf16 BsT[64][72];   // 64x64  (+8 pad), transposed

  const int tid  = threadIdx.x;
  const int lane = tid & 31;
  const int wave = tid >> 5;
  const int wm   = wave >> 1;         // 4 row chunks of 64
  const int wn   = wave & 1;          // 2 col chunks of 32
  const int m0   = blockIdx.y * 256;
  const int n0   = blockIdx.x * 64;
  const int lrow = lane & 15;
  const int hi   = lane >> 4;

  v8f acc[4][2];
#pragma unroll
  for (int i = 0; i < 4; ++i)
#pragma unroll
    for (int j = 0; j < 2; ++j)
#pragma unroll
      for (int e = 0; e < 8; ++e) acc[i][j][e] = 0.0f;

  const int brow = tid >> 2;          // 0..63
  const int bcol = (tid & 3) * 16;    // 0/16/32/48

  for (int k0 = 0; k0 < K; k0 += 64) {
    // ---- A tile 256x64 -> LDS ----
#if defined(HAVE_TDM)
    // One TDM per wave: rows [wave*32, wave*32+32) of the block tile.
    tdm_load_tile_2d(lds_offset(&As[wave * 32][0]),
                     A + (size_t)(m0 + wave * 32) * K + k0,
                     (unsigned)K, 32u, 64u, 32u, (unsigned)K);
#else
    {
      const __bf16* src = A + (size_t)(m0 + tid) * K + k0;
#pragma unroll
      for (int j = 0; j < 8; ++j)
        *reinterpret_cast<uint4*>(&As[tid][j * 8]) =
            *reinterpret_cast<const uint4*>(src + j * 8);
    }
#endif
    // ---- B tile 64x64 -> LDS transposed ----
    {
      const __bf16* bs = B + (size_t)(k0 + brow) * N + n0 + bcol;
      BF16x8 t0, t1;
      t0.q = *reinterpret_cast<const uint4*>(bs);
      t1.q = *reinterpret_cast<const uint4*>(bs + 8);
#pragma unroll
      for (int j = 0; j < 8; ++j) BsT[bcol + j][brow] = t0.h[j];
#pragma unroll
      for (int j = 0; j < 8; ++j) BsT[bcol + 8 + j][brow] = t1.h[j];
    }
#if defined(HAVE_TDM)
    __builtin_amdgcn_s_wait_tensorcnt((short)0);
#endif
    __syncthreads();

    const int c0  = hi * 8;           // A frag: K(e)=e+(e>=8?8:0)+hi*8
    const int kk0 = hi * 8;           // B frag: K(e)=e+(e>=8?8:0)+hi*8
#pragma unroll
    for (int ks = 0; ks < 2; ++ks) {  // two K=32 steps per slab
      v16bf af[4], bfr[2];
#pragma unroll
      for (int mi = 0; mi < 4; ++mi) {
        const int r = wm * 64 + mi * 16 + lrow;
        af[mi] = load_frag(&As[r][ks * 32 + c0], &As[r][ks * 32 + c0 + 16]);
      }
#pragma unroll
      for (int ni = 0; ni < 2; ++ni) {
        const int cidx = wn * 32 + ni * 16 + lrow;
        bfr[ni] = load_frag(&BsT[cidx][ks * 32 + kk0],
                            &BsT[cidx][ks * 32 + kk0 + 16]);
      }
#pragma unroll
      for (int mi = 0; mi < 4; ++mi)
#pragma unroll
        for (int ni = 0; ni < 2; ++ni)
          acc[mi][ni] = wmma_bf16(af[mi], bfr[ni], acc[mi][ni]);
    }
    __syncthreads();
  }

  // C/D layout: element r -> row r + hi*8, col = lrow
#pragma unroll
  for (int mi = 0; mi < 4; ++mi)
#pragma unroll
    for (int ni = 0; ni < 2; ++ni) {
      const int col = n0 + wn * 32 + ni * 16 + lrow;
#pragma unroll
      for (int r = 0; r < 8; ++r) {
        const int row = m0 + wm * 64 + mi * 16 + r + hi * 8;
        C[(size_t)row * N + col] = acc[mi][ni][r];
      }
    }
}

// ---------------------------------------------------------------------------
// RoPE + QKV split. qkv f32 [SEQ][73*64] ->
//   qb  bf16 [71][SEQ][64] (rotary), kb bf16 [SEQ][64] (rotary),
//   vT  bf16 [64][SEQ]     (transposed for contiguous P@V B-fragments)
// ---------------------------------------------------------------------------
__global__ void __launch_bounds__(256)
rope_split(const float* __restrict__ qkv, const float* __restrict__ cosT,
           const float* __restrict__ sinT, __bf16* __restrict__ qb,
           __bf16* __restrict__ kbuf, __bf16* __restrict__ vT) {
  const int s = blockIdx.x;
  const float* row = qkv + (size_t)s * QKV_N;
  for (int i = threadIdx.x; i < QKV_N; i += blockDim.x) {
    const int h = i >> 6;
    const int d = i & 63;
    const float x = row[i];
    if (h <= NHEADS) {
      const float other = row[(h << 6) + ((d < 32) ? d + 32 : d - 32)];
      const float rot   = (d < 32) ? -other : other;
      const float val = x * cosT[(s << 6) + d] + rot * sinT[(s << 6) + d];
      if (h < NHEADS)
        qb[((size_t)h * SEQ + s) * HDIM + d] = (__bf16)val;
      else
        kbuf[(size_t)s * HDIM + d] = (__bf16)val;
    } else {
      vT[(size_t)d * SEQ + s] = (__bf16)x;   // V transposed
    }
  }
}

// ---------------------------------------------------------------------------
// Flash attention, multi-query: K/V shared across 71 heads.
// Block = 4 waves; each wave owns 16 query rows of one head and streams
// causal key blocks of 32. Q fragments loaded once; softmax stats live in
// the WMMA C-register layout; P re-laid-out C->A through 1KB LDS per wave.
// ---------------------------------------------------------------------------
__global__ void __launch_bounds__(128)
attn_fwd(const __bf16* __restrict__ qb, const __bf16* __restrict__ kbuf,
         const __bf16* __restrict__ vT, __bf16* __restrict__ ctx) {
  __shared__ __align__(16) __bf16 Pb[4][16][40];

  const int head = blockIdx.y;
  const int lane = threadIdx.x & 31;
  const int wave = threadIdx.x >> 5;
  const int q0   = blockIdx.x * 64 + wave * 16;
  const int lrow = lane & 15;
  const int hi   = lane >> 4;
  const int roff = hi * 8;
  const int c0   = hi * 8;

  const __bf16* Q = qb + ((size_t)head * SEQ + q0 + lrow) * HDIM;
  v16bf aq[2];
#pragma unroll
  for (int s = 0; s < 2; ++s)
    aq[s] = load_frag(Q + s * 32 + c0, Q + s * 32 + c0 + 16);

  v8f o[4];
  float m[8], l[8];
#pragma unroll
  for (int ti = 0; ti < 4; ++ti)
#pragma unroll
    for (int e = 0; e < 8; ++e) o[ti][e] = 0.0f;
  const float NEG_BIG = -3.0e38f;
#pragma unroll
  for (int r = 0; r < 8; ++r) { m[r] = NEG_BIG; l[r] = 0.0f; }

  const float scale = 0.125f;          // 1/sqrt(64)
  const int kmax = q0 + 15;            // causal bound (wave-uniform)

  for (int kb0 = 0; kb0 <= kmax; kb0 += 32) {
    v8f sacc[2];
#pragma unroll
    for (int t = 0; t < 2; ++t) {
#pragma unroll
      for (int e = 0; e < 8; ++e) sacc[t][e] = 0.0f;
      const __bf16* Kp =
          kbuf + (size_t)(kb0 + t * 16 + lrow) * HDIM + hi * 8;
#pragma unroll
      for (int s = 0; s < 2; ++s) {
        v16bf bk = load_frag(Kp + s * 32, Kp + s * 32 + 16);
        sacc[t] = wmma_bf16(aq[s], bk, sacc[t]);
      }
    }

    float sc[2][8], mnew[8];
#pragma unroll
    for (int r = 0; r < 8; ++r) {
      const int qg = q0 + r + roff;
#pragma unroll
      for (int t = 0; t < 2; ++t) {
        const int key = kb0 + t * 16 + lrow;
        const float v = sacc[t][r] * scale;
        sc[t][r] = (key <= qg) ? v : NEG_BIG;
      }
      float v = fmaxf(sc[0][r], sc[1][r]);
      v = fmaxf(v, __shfl_xor(v, 1, 16));
      v = fmaxf(v, __shfl_xor(v, 2, 16));
      v = fmaxf(v, __shfl_xor(v, 4, 16));
      v = fmaxf(v, __shfl_xor(v, 8, 16));
      mnew[r] = fmaxf(m[r], v);
    }

#pragma unroll
    for (int r = 0; r < 8; ++r) {
      const float f = __expf(m[r] - mnew[r]);
      m[r] = mnew[r];
      l[r] *= f;
#pragma unroll
      for (int ti = 0; ti < 4; ++ti) o[ti][r] *= f;
      float rs = 0.0f;
#pragma unroll
      for (int t = 0; t < 2; ++t) {
        const float p = __expf(sc[t][r] - m[r]);
        rs += p;
        Pb[wave][r + roff][t * 16 + lrow] = (__bf16)p;
      }
      rs += __shfl_xor(rs, 1, 16);
      rs += __shfl_xor(rs, 2, 16);
      rs += __shfl_xor(rs, 4, 16);
      rs += __shfl_xor(rs, 8, 16);
      l[r] += rs;
    }

    asm volatile("s_wait_dscnt 0" ::: "memory");  // wave-local P RAW

    v16bf ap = load_frag(&Pb[wave][lrow][c0], &Pb[wave][lrow][c0 + 16]);
#pragma unroll
    for (int ti = 0; ti < 4; ++ti) {
      const __bf16* Vp = vT + (size_t)(ti * 16 + lrow) * SEQ + kb0 + hi * 8;
      v16bf bv = load_frag(Vp, Vp + 16);
      o[ti] = wmma_bf16(ap, bv, o[ti]);
    }
  }

#pragma unroll
  for (int ti = 0; ti < 4; ++ti) {
#pragma unroll
    for (int r = 0; r < 8; ++r) {
      const int row = q0 + r + roff;
      const int col = head * HDIM + ti * 16 + lrow;
      ctx[(size_t)row * HIDDEN + col] = (__bf16)(o[ti][r] / l[r]);
    }
  }
}

// ---------------------------------------------------------------------------
// Host-side orchestration (all on `stream`, graph-capture safe).
// ---------------------------------------------------------------------------
extern "C" void kernel_launch(void* const* d_in, const int* in_sizes, int n_in,
                              void* d_out, int out_size, void* d_ws,
                              size_t ws_size, hipStream_t stream) {
  (void)in_sizes; (void)n_in; (void)out_size; (void)ws_size;
  const float* x      = (const float*)d_in[0];   // [1,1,2048,4544]
  // d_in[1] = attention_mask: exactly causal -> implemented analytically
  const float* wqkv   = (const float*)d_in[2];   // [4544,4672]
  const float* wdense = (const float*)d_in[3];   // [4544,4544]
  const float* cosT   = (const float*)d_in[4];   // [2048,64]
  const float* sinT   = (const float*)d_in[5];   // [2048,64]
  float* out = (float*)d_out;                    // [2048,4544] f32

  char* ws = (char*)d_ws;
  size_t off = 0;
  auto take = [&](size_t bytes) -> char* {
    char* p = ws + off;
    off += (bytes + 255) & ~(size_t)255;
    return p;
  };
  __bf16* xb      = (__bf16*)take((size_t)SEQ * HIDDEN * 2);
  __bf16* wqkvb   = (__bf16*)take((size_t)HIDDEN * QKV_N * 2);
  __bf16* wdenseb = (__bf16*)take((size_t)HIDDEN * HIDDEN * 2);
  float*  qkv     = (float*) take((size_t)SEQ * QKV_N * 4);
  __bf16* qbuf    = (__bf16*)take((size_t)NHEADS * SEQ * HDIM * 2);
  __bf16* kbuf    = (__bf16*)take((size_t)SEQ * HDIM * 2);
  __bf16* vT      = (__bf16*)take((size_t)SEQ * HDIM * 2);
  __bf16* ctxb    = (__bf16*)take((size_t)SEQ * HIDDEN * 2);

  auto cvt = [&](const float* src, __bf16* dst, size_t n) {
    size_t n4 = n / 4;
    cvt_f32_bf16_v4<<<(unsigned)((n4 + 255) / 256), 256, 0, stream>>>(src, dst,
                                                                      n4);
  };
  cvt(x,      xb,      (size_t)SEQ * HIDDEN);
  cvt(wqkv,   wqkvb,   (size_t)HIDDEN * QKV_N);
  cvt(wdense, wdenseb, (size_t)HIDDEN * HIDDEN);

  // QKV GEMM: [2048,4544] @ [4544,4672] -> f32
  gemm_bf16_f32<<<dim3(QKV_N / 64, SEQ / 256), 256, 0, stream>>>(
      xb, wqkvb, qkv, SEQ, QKV_N, HIDDEN);

  // RoPE + split into Q(head-major) / K / V-transposed, all bf16
  rope_split<<<SEQ, 256, 0, stream>>>(qkv, cosT, sinT, qbuf, kbuf, vT);

  // Flash attention: grid = (query blocks of 64, heads)
  attn_fwd<<<dim3(SEQ / 64, NHEADS), 128, 0, stream>>>(qbuf, kbuf, vT, ctxb);

  // Output projection: [2048,4544] @ [4544,4544] -> f32 out
  gemm_bf16_f32<<<dim3(HIDDEN / 64, SEQ / 256), 256, 0, stream>>>(
      ctxb, wdenseb, out, SEQ, HIDDEN, HIDDEN);
}
